// PhaseHarmonicPrunedSeparated_24386824307128
// MI455X (gfx1250) — matrix operand order repaired and checked by reading, not compiled, not run on hardware
//
#include <hip/hip_runtime.h>
#include <hip/hip_bf16.h>
#include <math.h>

// ---------------------------------------------------------------------------
// PhaseHarmonicPrunedSeparated for MI455X (gfx1250, wave32)
//   B=4, C=1, T=32768, F=13 filters, K-modes {0,1,2} -> 39 harmonic rows
//   Strategy: FFT filter bank entirely in 320KB LDS; pruned pair correlations
//   recast as a dense complex Gram matrix computed with v_wmma_f32_16x16x32_f16.
// ---------------------------------------------------------------------------

typedef _Float16 v8h  __attribute__((ext_vector_type(8)));
typedef _Float16 v16h __attribute__((ext_vector_type(16)));
typedef float    v8f  __attribute__((ext_vector_type(8)));

#define T_LEN    32768
#define LOG2T    15
#define B_N      4
#define F_N      13
#define KMOD     3
#define ROWS_PAD 48           // 39 real rows padded to 3 WMMA tiles
#define NTHREADS 1024
#define SLICES   16
#define SLICE_LEN (T_LEN / SLICES)
#define PI_F 3.14159265358979323846f

__device__ __forceinline__ int brev15(int t) {
    return (int)(__brev((unsigned)t) >> (32 - LOG2T));
}

// ---------------------------------------------------------------------------
// Kernel 1: forward FFT of x[b] (32768-pt complex, whole transform in LDS)
// ---------------------------------------------------------------------------
__global__ void __launch_bounds__(NTHREADS)
fft_forward_kernel(const float* __restrict__ x, float* __restrict__ xhat)
{
    extern __shared__ float lds[];
    float* re = lds;
    float* im = lds + T_LEN;
    const int b   = blockIdx.x;
    const int tid = threadIdx.x;
    const float* xb = x + (size_t)b * T_LEN;

    for (int t = tid; t < T_LEN; t += NTHREADS) {
        const int d = brev15(t);
        re[d] = xb[t];
        im[d] = 0.0f;
    }
    __syncthreads();

    for (int s = 1; s <= LOG2T; ++s) {
        const int   half = 1 << (s - 1);
        const float wb   = -2.0f * PI_F / (float)(half << 1);   // forward sign
        for (int j = tid; j < (T_LEN >> 1); j += NTHREADS) {
            const int pos = j & (half - 1);
            const int i1  = ((j >> (s - 1)) << s) + pos;
            const int i2  = i1 + half;
            float sn, cs;
            __sincosf(wb * (float)pos, &sn, &cs);
            const float ar = re[i2], ai = im[i2];
            const float tr = cs * ar - sn * ai;
            const float ti = cs * ai + sn * ar;
            const float br = re[i1], bi = im[i1];
            re[i1] = br + tr; im[i1] = bi + ti;
            re[i2] = br - tr; im[i2] = bi - ti;
        }
        __syncthreads();
    }

    float* out = xhat + (size_t)b * T_LEN * 2;
    for (int t = tid; t < T_LEN; t += NTHREADS) {
        out[2 * t]     = re[t];
        out[2 * t + 1] = im[t];
    }
}

// ---------------------------------------------------------------------------
// Kernel 2: per (b,f): y = xhat * filt[f]; z = IFFT(y); emit harmonic rows
//   row rb+0: |z|        row rb+1: z        row rb+2: z^2/|z|
// as fp16 planes for the WMMA Gram stage, plus f32 per-row means.
// ---------------------------------------------------------------------------
__global__ void __launch_bounds__(NTHREADS)
filter_ifft_harm_kernel(const float* __restrict__ xhat,
                        const float* __restrict__ filt,
                        _Float16* __restrict__ Hre,
                        _Float16* __restrict__ Him,
                        float* __restrict__ Mean)
{
    extern __shared__ float lds[];
    float* re = lds;
    float* im = lds + T_LEN;
    const int f   = blockIdx.x;
    const int b   = blockIdx.y;
    const int tid = threadIdx.x;

    const float* xb = xhat + (size_t)b * T_LEN * 2;
    const float* fb = filt + (size_t)f * T_LEN * 2;

    // pointwise multiply in frequency domain, store bit-reversed for DIT IFFT
    for (int t = tid; t < T_LEN; t += NTHREADS) {
        const float xr = xb[2 * t], xi = xb[2 * t + 1];
        const float fr = fb[2 * t], fi = fb[2 * t + 1];
        const int d = brev15(t);
        re[d] = xr * fr - xi * fi;
        im[d] = xr * fi + xi * fr;
    }
    __syncthreads();

    // inverse FFT: positive-sign twiddles; 1/T scaling applied below
    for (int s = 1; s <= LOG2T; ++s) {
        const int   half = 1 << (s - 1);
        const float wb   = 2.0f * PI_F / (float)(half << 1);
        for (int j = tid; j < (T_LEN >> 1); j += NTHREADS) {
            const int pos = j & (half - 1);
            const int i1  = ((j >> (s - 1)) << s) + pos;
            const int i2  = i1 + half;
            float sn, cs;
            __sincosf(wb * (float)pos, &sn, &cs);
            const float ar = re[i2], ai = im[i2];
            const float tr = cs * ar - sn * ai;
            const float ti = cs * ai + sn * ar;
            const float br = re[i1], bi = im[i1];
            re[i1] = br + tr; im[i1] = bi + ti;
            re[i2] = br - tr; im[i2] = bi - ti;
        }
        __syncthreads();
    }

    const float invT = 1.0f / (float)T_LEN;
    const int   rb   = f * KMOD;
    _Float16* h0r = Hre + (size_t)(b * ROWS_PAD + rb + 0) * T_LEN;
    _Float16* h0i = Him + (size_t)(b * ROWS_PAD + rb + 0) * T_LEN;
    _Float16* h1r = Hre + (size_t)(b * ROWS_PAD + rb + 1) * T_LEN;
    _Float16* h1i = Him + (size_t)(b * ROWS_PAD + rb + 1) * T_LEN;
    _Float16* h2r = Hre + (size_t)(b * ROWS_PAD + rb + 2) * T_LEN;
    _Float16* h2i = Him + (size_t)(b * ROWS_PAD + rb + 2) * T_LEN;

    float s0 = 0.f, s1r = 0.f, s1i = 0.f, s2r = 0.f, s2i = 0.f;
    for (int t = tid; t < T_LEN; t += NTHREADS) {
        const float zr = re[t] * invT;
        const float zi = im[t] * invT;
        const float r  = sqrtf(zr * zr + zi * zi);
        const float ir = (r > 0.f) ? (1.0f / r) : 0.0f;
        const float e2r = (zr * zr - zi * zi) * ir;   // |z| e^{2i theta}
        const float e2i = (2.0f * zr * zi) * ir;
        h0r[t] = (_Float16)r;   h0i[t] = (_Float16)0.f;
        h1r[t] = (_Float16)zr;  h1i[t] = (_Float16)zi;
        h2r[t] = (_Float16)e2r; h2i[t] = (_Float16)e2i;
        s0 += r; s1r += zr; s1i += zi; s2r += e2r; s2i += e2i;
    }
    __syncthreads();  // done reading FFT result; re[] reused for reduction

    float vals[5] = { s0, s1r, s1i, s2r, s2i };
    float tot[5];
    for (int q = 0; q < 5; ++q) {
        re[tid] = vals[q];
        __syncthreads();
        for (int off = NTHREADS >> 1; off > 0; off >>= 1) {
            if (tid < off) re[tid] += re[tid + off];
            __syncthreads();
        }
        tot[q] = re[0];
        __syncthreads();
    }
    if (tid == 0) {
        float* M = Mean + (size_t)(b * ROWS_PAD + rb) * 2;
        M[0] = tot[0] * invT; M[1] = 0.0f;        // k=0 mean (real)
        M[2] = tot[1] * invT; M[3] = tot[2] * invT;
        M[4] = tot[3] * invT; M[5] = tot[4] * invT;
    }
}

// ---------------------------------------------------------------------------
// Kernel 3: complex Gram G[b] = H[b] * H[b]^H via v_wmma_f32_16x16x32_f16.
// One wave per (b, 16x16 tile, T-slice); accumulate in f32, atomic combine.
//   Re(G)  = Ar*Br + Ai*Bi
//   Im(G)  = Ai*Br - Ar*Bi   -> kept as two accumulators (imP, imN) so the
//                               hot loop has ZERO per-iteration VALU work;
//                               single v_sub combine after the K loop.
// ---------------------------------------------------------------------------
__global__ void __launch_bounds__(32)
gram_wmma_kernel(const _Float16* __restrict__ Hre,
                 const _Float16* __restrict__ Him,
                 float* __restrict__ G)
{
    const int tile  = blockIdx.x;            // 0..8 -> (ti, tj)
    const int ti    = tile / 3;
    const int tj    = tile % 3;
    const int slice = blockIdx.y;            // 0..SLICES-1
    const int b     = blockIdx.z;
    const int lane  = threadIdx.x & 31;
    const int nm    = lane & 15;             // row within tile (A) / col (B)
    const int hi    = lane >> 4;             // K half-selector (wave32 layout)

    const size_t rowA = (size_t)(b * ROWS_PAD + ti * 16 + nm) * T_LEN;
    const size_t rowB = (size_t)(b * ROWS_PAD + tj * 16 + nm) * T_LEN;
    const _Float16* ar_p = Hre + rowA;
    const _Float16* ai_p = Him + rowA;
    const _Float16* br_p = Hre + rowB;
    const _Float16* bi_p = Him + rowB;

    v8f acc_re  = {};   // Ar*Br + Ai*Bi
    v8f acc_imP = {};   // Ai*Br
    v8f acc_imN = {};   // Ar*Bi   (subtracted after the loop)

    const int t0 = slice * SLICE_LEN;
    for (int t = t0; t < t0 + SLICE_LEN; t += 32) {
        // A fragment (16x32 f16): lane holds 16 contiguous K at t + 16*hi
        const int aOff = t + 16 * hi;
        const v8h ar_lo = *(const v8h*)(ar_p + aOff);
        const v8h ar_hi = *(const v8h*)(ar_p + aOff + 8);
        const v8h ai_lo = *(const v8h*)(ai_p + aOff);
        const v8h ai_hi = *(const v8h*)(ai_p + aOff + 8);
        const v16h Ar = __builtin_shufflevector(ar_lo, ar_hi,
            0,1,2,3,4,5,6,7,8,9,10,11,12,13,14,15);
        const v16h Ai = __builtin_shufflevector(ai_lo, ai_hi,
            0,1,2,3,4,5,6,7,8,9,10,11,12,13,14,15);

        // B fragment (32x16 f16): lane holds 16 contiguous K at t + 16*hi
        const int bOff = t + 16 * hi;
        const v16h Br  = *(const v16h*)(br_p + bOff);
        const v16h Bi  = *(const v16h*)(bi_p + bOff);

        acc_re  = __builtin_amdgcn_wmma_f32_16x16x32_f16(
            false, Ar, false, Br, (short)0, acc_re,  false, false);
        acc_re  = __builtin_amdgcn_wmma_f32_16x16x32_f16(
            false, Ai, false, Bi, (short)0, acc_re,  false, false);
        acc_imP = __builtin_amdgcn_wmma_f32_16x16x32_f16(
            false, Ai, false, Br, (short)0, acc_imP, false, false);
        acc_imN = __builtin_amdgcn_wmma_f32_16x16x32_f16(
            false, Ar, false, Bi, (short)0, acc_imN, false, false);

        // stage next K-chunk into cache (global_prefetch_b8 on gfx1250)
        __builtin_prefetch(ar_p + aOff + 32, 0, 3);
        __builtin_prefetch(br_p + bOff + 32, 0, 3);
    }

    const v8f acc_im = acc_imP - acc_imN;    // one-time combine (8 v_sub_f32)

    // D layout: VGPR v, lane l -> M = v + 8*(l>>4), N = l&15
    for (int v = 0; v < 8; ++v) {
        const int m = ti * 16 + v + 8 * hi;
        const int n = tj * 16 + nm;
        float* g = G + ((size_t)(b * ROWS_PAD + m) * ROWS_PAD + n) * 2;
        atomicAdd(g + 0, acc_re[v]);
        atomicAdd(g + 1, acc_im[v]);
    }
}

// ---------------------------------------------------------------------------
// Kernel 4: gather pruned pairs, subtract mean product, emit (re, im)
// ---------------------------------------------------------------------------
__global__ void finalize_kernel(const int* __restrict__ xi_idx,
                                const int* __restrict__ ks,
                                const float* __restrict__ G,
                                const float* __restrict__ Mean,
                                float* __restrict__ out, int P)
{
    const int b = blockIdx.y;
    const int p = blockIdx.x * blockDim.x + threadIdx.x;
    if (p >= P) return;
    const int r1 = xi_idx[2 * p]     * KMOD + ks[2 * p];
    const int r2 = xi_idx[2 * p + 1] * KMOD + ks[2 * p + 1];
    const float invT = 1.0f / (float)T_LEN;
    const float* g  = G + ((size_t)(b * ROWS_PAD + r1) * ROWS_PAD + r2) * 2;
    const float* m1 = Mean + (size_t)(b * ROWS_PAD + r1) * 2;
    const float* m2 = Mean + (size_t)(b * ROWS_PAD + r2) * 2;
    const float gre = g[0] * invT - (m1[0] * m2[0] + m1[1] * m2[1]);
    const float gim = g[1] * invT - (m1[1] * m2[0] - m1[0] * m2[1]);
    float* o = out + (size_t)(b * P + p) * 2;
    o[0] = gre;
    o[1] = gim;
}

// ---------------------------------------------------------------------------
extern "C" void kernel_launch(void* const* d_in, const int* in_sizes, int n_in,
                              void* d_out, int out_size, void* d_ws, size_t ws_size,
                              hipStream_t stream) {
    const float* x    = (const float*)d_in[0];   // (B,C,T) f32
    const float* filt = (const float*)d_in[1];   // (F,T,2) f32
    const int*   xi   = (const int*)d_in[2];     // (P,2) i32
    const int*   kk   = (const int*)d_in[3];     // (P,2) i32
    float*       out  = (float*)d_out;           // (B,C,P,2) f32
    const int P = in_sizes[2] / 2;

    // workspace layout
    char* ws = (char*)d_ws;
    const size_t Hplane    = (size_t)B_N * ROWS_PAD * T_LEN;       // halves
    _Float16* Hre  = (_Float16*)(ws);
    _Float16* Him  = (_Float16*)(ws + Hplane * sizeof(_Float16));
    float*    xhat = (float*)(ws + 2 * Hplane * sizeof(_Float16));
    const size_t xhatBytes = (size_t)B_N * T_LEN * 2 * sizeof(float);
    float*    G    = (float*)((char*)xhat + xhatBytes);
    const size_t Gbytes    = (size_t)B_N * ROWS_PAD * ROWS_PAD * 2 * sizeof(float);
    float*    Mean = (float*)((char*)G + Gbytes);
    const size_t Mbytes    = (size_t)B_N * ROWS_PAD * 2 * sizeof(float);
    const size_t totalBytes = 2 * Hplane * sizeof(_Float16) + xhatBytes + Gbytes + Mbytes;

    // zero pad-rows of H and the atomic accumulators (every call)
    hipMemsetAsync(d_ws, 0, totalBytes, stream);

    const size_t ldsBytes = 2 * (size_t)T_LEN * sizeof(float);     // 256 KB of 320 KB

    fft_forward_kernel<<<dim3(B_N), dim3(NTHREADS), ldsBytes, stream>>>(x, xhat);

    filter_ifft_harm_kernel<<<dim3(F_N, B_N), dim3(NTHREADS), ldsBytes, stream>>>(
        xhat, filt, Hre, Him, Mean);

    gram_wmma_kernel<<<dim3(9, SLICES, B_N), dim3(32), 0, stream>>>(Hre, Him, G);

    finalize_kernel<<<dim3((P + 255) / 256, B_N), dim3(256), 0, stream>>>(
        xi, kk, G, Mean, out, P);
}
